// SparseResidualBlock_3968549781705
// MI455X (gfx1250) — matrix operand order repaired and checked by reading, not compiled, not run on hardware
//
#include <hip/hip_runtime.h>

// ---------------------------------------------------------------------------
// CDNA5 / gfx1250 sparse residual block.
// bf16 WMMA (v_wmma_f32_16x16x32_bf16) with fp32 accumulate.
// Depth-3 rotating register pipeline on B fragments (4 live buffers), a
// one-chunk-ahead A pipeline, and sched_barrier(0) fences so the scheduler
// cannot sink the loads back onto their consuming WMMAs.
// ---------------------------------------------------------------------------

typedef __attribute__((ext_vector_type(16))) __bf16        v16bf;
typedef __attribute__((ext_vector_type(8)))  float         v8f;
typedef __attribute__((ext_vector_type(4)))  unsigned int  u32x4;

union FragU { u32x4 q[2]; v16bf v; };

namespace cfg {
  constexpr int N     = 100000;   // sites
  constexpr int KNB   = 27;       // neighbor offsets
  constexpr int C1    = 64;       // input channels
  constexpr int D     = 128;      // output channels (both convs)
  constexpr int TILES = N / 16;   // 6250 (N % 16 == 0 -> all tiles full)
  constexpr int WPB   = 8;        // waves per block (256 threads)
  constexpr int CONV_BLOCKS = (TILES + WPB - 1) / WPB;   // 782
  constexpr int BPIPE = 3;        // B fragments in flight (4 buffers)
}

static __device__ __forceinline__ unsigned short f2bf(float f) {
  // round-to-nearest-even f32 -> bf16 (inputs are finite)
  unsigned int u = __float_as_uint(f);
  u += 0x7fffu + ((u >> 16) & 1u);
  return (unsigned short)(u >> 16);
}

static __device__ __forceinline__ void loadB(FragU& b, const unsigned short* p) {
  const u32x4* bp = (const u32x4*)p;
  b.q[0] = bp[0];
  b.q[1] = bp[1];
}

// 16-bit A 16x32 fragment: lo half-wave lanes hold K{0..15} of row m,
// hi half-wave lanes hold K{16..31}  ==  two 16B loads per lane.
static __device__ __forceinline__ void loadA(FragU& a, const unsigned short* arow,
                                             int kc, int hi) {
  a.q[0] = *(const u32x4*)(arow + kc * 32 + hi * 16);
  a.q[1] = *(const u32x4*)(arow + kc * 32 + hi * 16 + 8);
}

// --------------------------- prep kernels ----------------------------------

__global__ void k_init(unsigned short* __restrict__ xpad,   // 64 bf16 (sentinel row of x)
                       unsigned short* __restrict__ hpad,   // 128 bf16 (sentinel row of h1)
                       float* __restrict__ stats) {         // 512 f32 (sum1,sq1,sum2,sq2)
  int t = threadIdx.x;
  if (t < 64)  xpad[t] = 0;
  if (t < 128) hpad[t] = 0;
  stats[t]       = 0.f;
  stats[256 + t] = 0.f;
}

__global__ void k_cvt_bf16(const float* __restrict__ src,
                           unsigned short* __restrict__ dst, int n) {
  int i = blockIdx.x * 256 + threadIdx.x;
  if (i < n) dst[i] = f2bf(src[i]);
}

// Swizzle W[k][c][d] (f32) into per-lane B-fragment order (bf16):
// fragment (k, kc=c/32, dc=d/16) of 512 elems laid out [lane][e]:
//   lane = (d&15) + (c&16 ? 16 : 0) ; e = c&15  (contiguous k order, matches loadA)
__global__ void k_swizzle_w(const float* __restrict__ src,
                            unsigned short* __restrict__ dst,
                            int KC, int CD, int Dd, int total) {
  int i = blockIdx.x * 256 + threadIdx.x;
  if (i >= total) return;
  int k   = i / CD;
  int rem = i - k * CD;
  int c   = rem / Dd;
  int d   = rem - c * Dd;
  int kc  = c >> 5, kk = c & 31, dc = d >> 4, n = d & 15;
  int lane = n + (kk & 16);
  int e    = kk & 15;
  size_t di = ((((size_t)k * KC + kc) * 8 + dc) * 512) + (size_t)lane * 16 + e;
  dst[di] = f2bf(src[i]);
}

// --------------------------- conv + stats ----------------------------------
// One wave = 16-row M-tile x 128 output columns (8 accumulator tiles).
// A fragments are gathered straight from global (L2-resident activation
// table). B fragments stream from the contiguous swizzled weight table:
// fragment addresses are uniform in the global step index, so the rotating
// pipeline needs no branch at the neighbor-offset boundary. sched_barrier(0)
// after each step's load issue keeps the pipeline materialized.

template<int CIN>
__global__ void __launch_bounds__(256) k_conv_wmma(
    const unsigned short* __restrict__ act,   // (N+1) x CIN bf16 (row N zero)
    const int*            __restrict__ nbr,   // N x 27
    const unsigned short* __restrict__ wsw,   // 27 x KC x 8 x 512 bf16 (+4KB pad)
    float*                __restrict__ yout,  // N x 128 f32
    float* __restrict__ gsum, float* __restrict__ gsq) {
  constexpr int KC = CIN / 32;
  constexpr int JN = KC * 8;                  // wmma steps per neighbor offset
  __shared__ float sred[256];
  int t = threadIdx.x;
  sred[t] = 0.f;
  __syncthreads();

  int lane = t & 31;
  int wave = t >> 5;
  int tile = blockIdx.x * cfg::WPB + wave;
  if (tile < cfg::TILES) {            // wave-uniform: EXEC stays all-ones
    int m0 = tile * 16;
    int m  = lane & 15;
    int hi = lane >> 4;               // half-wave select
    v8f acc[8];
    v8f vz = {0.f, 0.f, 0.f, 0.f, 0.f, 0.f, 0.f, 0.f};
#pragma unroll
    for (int dc = 0; dc < 8; ++dc) acc[dc] = vz;

    const int* nrow = nbr + (size_t)(m0 + m) * cfg::KNB;

    FragU a[2], b[4];
    // ---- pipeline preamble: fragments 0..BPIPE-1 and A chunk 0 in flight
    int row0 = nrow[0];
    int rowNext = nrow[1];
    const unsigned short* arow = act + (size_t)row0 * CIN;
    const unsigned short* wkl  = wsw + lane * 16;   // + (k*JN + frag)*512
    loadA(a[0], arow, 0, hi);
    loadB(b[0], wkl + 0 * 512);
    loadB(b[1], wkl + 1 * 512);
    loadB(b[2], wkl + 2 * 512);

    for (int k = 0; k < cfg::KNB; ++k) {
      const unsigned short* wk    = wkl + (size_t)k * JN * 512;
      const unsigned short* arowN = act + (size_t)rowNext * CIN; // k+1's row
      rowNext = (k + 2 < cfg::KNB) ? nrow[k + 2] : 0;
#pragma unroll
      for (int j = 0; j < JN; ++j) {
        // B: issue fragment j+BPIPE (contiguous across the k boundary; the
        // weight table has a tail pad so the last overshoots stay mapped)
        loadB(b[(j + cfg::BPIPE) & 3], wk + (j + cfg::BPIPE) * 512);
        // A: issue next chunk a full 8 steps ahead
        if ((j & 7) == 0) {
          int fA = j + 8;
          if (fA < JN) loadA(a[(fA >> 3) & 1], arow,  fA >> 3, hi);
          else         loadA(a[(fA >> 3) & 1], arowN, 0,       hi);
        }
        // fence: loads above may not sink below, wmma may not hoist above
        __builtin_amdgcn_sched_barrier(0);
        acc[j & 7] = __builtin_amdgcn_wmma_f32_16x16x32_bf16(
            false, a[(j >> 3) & 1].v, false, b[j & 3].v,
            (short)0, acc[j & 7], false, false);
      }
      arow = arowN;
    }

    // store y and accumulate per-column sum / sum^2 (C/D layout: VGPR r ->
    // row m0+r+hi*8, col dc*16 + lane&15)
#pragma unroll
    for (int dc = 0; dc < 8; ++dc) {
      int col = dc * 16 + m;
      float s = 0.f, s2 = 0.f;
#pragma unroll
      for (int r8 = 0; r8 < 8; ++r8) {
        float v = acc[dc][r8];
        yout[(size_t)(m0 + r8 + hi * 8) * cfg::D + col] = v;
        s  += v;
        s2 += v * v;
      }
      atomicAdd(&sred[col],       s);
      atomicAdd(&sred[128 + col], s2);
    }
  }
  __syncthreads();
  if (t < 128) atomicAdd(&gsum[t],      sred[t]);
  else         atomicAdd(&gsq[t - 128], sred[t]);
}

// --------------------------- BN finalize / apply ---------------------------

__global__ void k_finalize(const float* __restrict__ sum, const float* __restrict__ sq,
                           const float* __restrict__ gamma, const float* __restrict__ beta,
                           float* __restrict__ scale, float* __restrict__ bias) {
  int c = threadIdx.x;
  if (c < 128) {
    const float invN = 1.0f / (float)cfg::N;
    float mu  = sum[c] * invN;
    float var = sq[c] * invN - mu * mu;
    float s   = gamma[c] * rsqrtf(var + 1e-5f);
    scale[c] = s;
    bias[c]  = beta[c] - mu * s;
  }
}

__global__ void k_bnrelu_bf16(const float* __restrict__ y,
                              const float* __restrict__ scale,
                              const float* __restrict__ bias,
                              unsigned short* __restrict__ h, int n) {
  int i = blockIdx.x * 256 + threadIdx.x;
  if (i < n) {
    int col = i & 127;
    float v = y[i] * scale[col] + bias[col];
    h[i] = f2bf(v > 0.f ? v : 0.f);
  }
}

// ------------------- fused BN2+ReLU + skip GEMM + output -------------------
// acc initialized with relu(y2*scale+bias) in the C/D layout, then the skip
// GEMM (x_bf16 @ W_skip) accumulates on top via WMMA; store fp32 out.

__global__ void __launch_bounds__(256) k_final_skip(
    const float* __restrict__ y2, const float* __restrict__ scale,
    const float* __restrict__ bias,
    const unsigned short* __restrict__ xbf,  // (N+1) x 64 bf16
    const unsigned short* __restrict__ wsk,  // 2 x 8 x 512 bf16 (+4KB pad)
    float* __restrict__ out) {
  int t = threadIdx.x;
  int lane = t & 31;
  int wave = t >> 5;
  int tile = blockIdx.x * cfg::WPB + wave;
  if (tile >= cfg::TILES) return;
  int m0 = tile * 16;
  int m  = lane & 15;
  int hi = lane >> 4;

  // issue both A fragments and the first BPIPE B fragments up front
  const unsigned short* arow = xbf + (size_t)(m0 + m) * cfg::C1;
  const unsigned short* wl   = wsk + lane * 16;
  FragU a[2], b[4];
  loadA(a[0], arow, 0, hi);
  loadA(a[1], arow, 1, hi);
  loadB(b[0], wl + 0 * 512);
  loadB(b[1], wl + 1 * 512);
  loadB(b[2], wl + 2 * 512);

  v8f acc[8];
#pragma unroll
  for (int dc = 0; dc < 8; ++dc) {
    int col = dc * 16 + m;
    float sc = scale[col], bi = bias[col];
#pragma unroll
    for (int r8 = 0; r8 < 8; ++r8) {
      float v = y2[(size_t)(m0 + r8 + hi * 8) * cfg::D + col] * sc + bi;
      acc[dc][r8] = v > 0.f ? v : 0.f;
    }
  }

#pragma unroll
  for (int j = 0; j < 16; ++j) {
    loadB(b[(j + cfg::BPIPE) & 3], wl + (j + cfg::BPIPE) * 512); // tail pad keeps overshoot mapped
    __builtin_amdgcn_sched_barrier(0);
    acc[j & 7] = __builtin_amdgcn_wmma_f32_16x16x32_bf16(
        false, a[(j >> 3) & 1].v, false, b[j & 3].v,
        (short)0, acc[j & 7], false, false);
  }

#pragma unroll
  for (int dc = 0; dc < 8; ++dc) {
    int col = dc * 16 + m;
#pragma unroll
    for (int r8 = 0; r8 < 8; ++r8)
      out[(size_t)(m0 + r8 + hi * 8) * cfg::D + col] = acc[dc][r8];
  }
}

// ------------------------------- launcher ----------------------------------

extern "C" void kernel_launch(void* const* d_in, const int* in_sizes, int n_in,
                              void* d_out, int out_size, void* d_ws, size_t ws_size,
                              hipStream_t stream) {
  (void)in_sizes; (void)n_in; (void)out_size; (void)ws_size;
  using namespace cfg;
  const float* x     = (const float*)d_in[0];
  const int*   nbr   = (const int*)  d_in[1];
  const float* W1    = (const float*)d_in[2];
  const float* g1    = (const float*)d_in[3];
  const float* b1    = (const float*)d_in[4];
  const float* W2    = (const float*)d_in[5];
  const float* g2    = (const float*)d_in[6];
  const float* b2    = (const float*)d_in[7];
  const float* Wsk   = (const float*)d_in[8];
  float*       out   = (float*)d_out;

  unsigned char* ws = (unsigned char*)d_ws;
  size_t off = 0;
  auto alloc = [&](size_t bytes) -> void* {
    void* p = ws + off;
    off = (off + bytes + 255) & ~(size_t)255;
    return p;
  };
  // weight tables get a 4KB tail pad: the rotating pipeline overshoots by up
  // to BPIPE fragments (3KB) on the last neighbor offset (values unused).
  unsigned short* XBF  = (unsigned short*)alloc((size_t)(N + 1) * C1 * 2);
  unsigned short* W1S  = (unsigned short*)alloc((size_t)KNB * 2 * 8 * 512 * 2 + 4096);
  unsigned short* W2S  = (unsigned short*)alloc((size_t)KNB * 4 * 8 * 512 * 2 + 4096);
  unsigned short* WSS  = (unsigned short*)alloc((size_t)2 * 8 * 512 * 2 + 4096);
  float*          Y    = (float*)alloc((size_t)N * D * 4);        // y1 then y2
  unsigned short* H1   = (unsigned short*)alloc((size_t)(N + 1) * D * 2);
  float*          STAT = (float*)alloc(512 * 4);   // sum1, sq1, sum2, sq2
  float*          SB   = (float*)alloc(512 * 4);   // scale1, bias1, scale2, bias2
  float* SUM1 = STAT,       *SQ1 = STAT + 128, *SUM2 = STAT + 256, *SQ2 = STAT + 384;
  float* SC1  = SB,         *BI1 = SB + 128,   *SC2  = SB + 256,   *BI2 = SB + 384;

  // prep
  k_init<<<1, 256, 0, stream>>>(XBF + (size_t)N * C1, H1 + (size_t)N * D, STAT);
  {
    int n = N * C1;
    k_cvt_bf16<<<(n + 255) / 256, 256, 0, stream>>>(x, XBF, n);
  }
  k_swizzle_w<<<(KNB * C1 * D + 255) / 256, 256, 0, stream>>>(W1, W1S, 2, C1 * D, D, KNB * C1 * D);
  k_swizzle_w<<<(KNB * D * D + 255) / 256, 256, 0, stream>>>(W2, W2S, 4, D * D, D, KNB * D * D);
  k_swizzle_w<<<(C1 * D + 255) / 256, 256, 0, stream>>>(Wsk, WSS, 2, C1 * D, D, C1 * D);

  // conv1 + stats -> BN1 -> h1 (bf16)
  k_conv_wmma<C1><<<CONV_BLOCKS, 256, 0, stream>>>(XBF, nbr, W1S, Y, SUM1, SQ1);
  k_finalize<<<1, 128, 0, stream>>>(SUM1, SQ1, g1, b1, SC1, BI1);
  {
    int n = N * D;
    k_bnrelu_bf16<<<(n + 255) / 256, 256, 0, stream>>>(Y, SC1, BI1, H1, n);
  }

  // conv2 + stats -> BN2 fused with skip GEMM + residual add
  k_conv_wmma<D><<<CONV_BLOCKS, 256, 0, stream>>>(H1, nbr, W2S, Y, SUM2, SQ2);
  k_finalize<<<1, 128, 0, stream>>>(SUM2, SQ2, g2, b2, SC2, BI2);
  k_final_skip<<<CONV_BLOCKS, 256, 0, stream>>>(Y, SC2, BI2, XBF, WSS, out);
}
